// BitLinear_6047313952927
// MI455X (gfx1250) — 8-bit matrix operand order repaired and checked by reading, compiled, not run on hardware
//
#include <hip/hip_runtime.h>
#include <cstdint>
#include <cstddef>


typedef int v8i __attribute__((ext_vector_type(8)));

__global__ __launch_bounds__(256) void k_abs_partial(const float* __restrict__ w,
                                                     float* __restrict__ partials,
                                                     int n4) {
    __shared__ float red[256];
    const float4* w4 = (const float4*)w;
    float s = 0.0f;
    for (int i = blockIdx.x * blockDim.x + threadIdx.x; i < n4; i += gridDim.x * blockDim.x) {
        float4 v = w4[i];
        s += fabsf(v.x) + fabsf(v.y) + fabsf(v.z) + fabsf(v.w);
    }
    red[threadIdx.x] = s;
    __syncthreads();
    for (int off = 128; off > 0; off >>= 1) {
        if (threadIdx.x < off) red[threadIdx.x] += red[threadIdx.x + off];
        __syncthreads();
    }
    if (threadIdx.x == 0) partials[blockIdx.x] = red[0];
}

__global__ __launch_bounds__(256) void k_wscale(const float* __restrict__ partials,
                                                float* __restrict__ scalars,
                                                int nPart, float invCount) {
    __shared__ float red[256];
    float s = 0.0f;
    for (int i = threadIdx.x; i < nPart; i += 256) s += partials[i];
    red[threadIdx.x] = s;
    __syncthreads();
    for (int off = 128; off > 0; off >>= 1) {
        if (threadIdx.x < off) red[threadIdx.x] += red[threadIdx.x + off];
        __syncthreads();
    }
    if (threadIdx.x == 0) {
        float mean = red[0] * invCount;
        scalars[0] = 1.0f / fmaxf(mean, 1e-5f);
    }
}

__global__ __launch_bounds__(256) void k_quant_w(const float* __restrict__ w,
                                                 const float* __restrict__ scalars,
                                                 signed char* __restrict__ wq,
                                                 int n4) {
    const float ws = scalars[0];
    const float4* w4 = (const float4*)w;
    int* out4 = (int*)wq;
    for (int i = blockIdx.x * blockDim.x + threadIdx.x; i < n4; i += gridDim.x * blockDim.x) {
        float4 v = w4[i];
        int b0 = (int)fminf(fmaxf(rintf(v.x * ws), -1.0f), 1.0f);
        int b1 = (int)fminf(fmaxf(rintf(v.y * ws), -1.0f), 1.0f);
        int b2 = (int)fminf(fmaxf(rintf(v.z * ws), -1.0f), 1.0f);
        int b3 = (int)fminf(fmaxf(rintf(v.w * ws), -1.0f), 1.0f);
        out4[i] = (b0 & 0xff) | ((b1 & 0xff) << 8) | ((b2 & 0xff) << 16) | ((b3 & 0xff) << 24);
    }
}

__global__ __launch_bounds__(256) void k_quant_x(const float* __restrict__ x,
                                                 const float* __restrict__ g,
                                                 const float* __restrict__ scalars,
                                                 signed char* __restrict__ xq,
                                                 float* __restrict__ dq,
                                                 int D) {
    __shared__ float red[256];
    const int t = blockIdx.x;
    const int tid = threadIdx.x;
    const float4* xr = (const float4*)(x + (size_t)t * D);
    const float4* g4 = (const float4*)g;

    float4 xv[4], gv[4];
    float ssq = 0.0f;
#pragma unroll
    for (int i = 0; i < 4; ++i) {
        int idx = i * 256 + tid;
        xv[i] = xr[idx];
        gv[i] = g4[idx];
        ssq += xv[i].x * xv[i].x + xv[i].y * xv[i].y + xv[i].z * xv[i].z + xv[i].w * xv[i].w;
    }
    red[tid] = ssq;
    __syncthreads();
    for (int off = 128; off > 0; off >>= 1) {
        if (tid < off) red[tid] += red[tid + off];
        __syncthreads();
    }
    const float blockSum = red[0];
    __syncthreads();

    const float r = rsqrtf(blockSum / (float)D + 1e-5f);
    float mx = 0.0f;
#pragma unroll
    for (int i = 0; i < 4; ++i) {
        xv[i].x *= r * gv[i].x;
        xv[i].y *= r * gv[i].y;
        xv[i].z *= r * gv[i].z;
        xv[i].w *= r * gv[i].w;
        mx = fmaxf(mx, fmaxf(fmaxf(fabsf(xv[i].x), fabsf(xv[i].y)),
                             fmaxf(fabsf(xv[i].z), fabsf(xv[i].w))));
    }
    red[tid] = mx;
    __syncthreads();
    for (int off = 128; off > 0; off >>= 1) {
        if (tid < off) red[tid] = fmaxf(red[tid], red[tid + off]);
        __syncthreads();
    }
    const float xscale = 127.0f / fmaxf(red[0], 1e-5f);

    int* outw = (int*)(xq + (size_t)t * D);
#pragma unroll
    for (int i = 0; i < 4; ++i) {
        int idx = i * 256 + tid;
        int b0 = (int)fminf(fmaxf(rintf(xv[i].x * xscale), -128.0f), 127.0f);
        int b1 = (int)fminf(fmaxf(rintf(xv[i].y * xscale), -128.0f), 127.0f);
        int b2 = (int)fminf(fmaxf(rintf(xv[i].z * xscale), -128.0f), 127.0f);
        int b3 = (int)fminf(fmaxf(rintf(xv[i].w * xscale), -128.0f), 127.0f);
        outw[idx] = (b0 & 0xff) | ((b1 & 0xff) << 8) | ((b2 & 0xff) << 16) | ((b3 & 0xff) << 24);
    }
    if (tid == 0) dq[t] = 1.0f / (scalars[0] * xscale);
}

__device__ __forceinline__ uint32_t lds_off32(const void* p) {
    return (uint32_t)(uintptr_t)p;
}

__device__ __forceinline__ void async_copy_b128(uint32_t lds_dst, uint64_t sbase, uint32_t voff) {
    asm volatile("global_load_async_to_lds_b128 %0, %1, %2"
                 :: "v"(lds_dst), "v"(voff), "s"(sbase)
                 : "memory");
}

__device__ __forceinline__ void wait_async0() {
#if __has_builtin(__builtin_amdgcn_s_wait_asynccnt)
    __builtin_amdgcn_s_wait_asynccnt(0);
#else
    asm volatile("s_wait_asynccnt 0" ::: "memory");
#endif
}

__global__ __launch_bounds__(256) void k_gemm(const signed char* __restrict__ xq,
                                              const signed char* __restrict__ wq,
                                              const float* __restrict__ dq,
                                              float* __restrict__ out,
                                              int D, int N) {
    __shared__ __align__(16) signed char lA[2][128 * 64];
    __shared__ __align__(16) signed char lB[2][128 * 64];
    __shared__ float ldq[128];

    const int tid = threadIdx.x;
    const int lane = tid & 31;
    const int wid = tid >> 5;
    const int wm = (wid & 3) * 32;
    const int wn = (wid >> 2) * 64;
    const int t0 = blockIdx.y * 128;
    const int n0 = blockIdx.x * 128;

    const int rowA = tid >> 1;
    const int segA = tid & 1;
    const int swzA = (rowA >> 2) & 3;
    const int swzB = ((rowA >> 2) & 1) << 1;

    uint32_t dstA0[2], dstA1[2], dstB0[2], dstB1[2];
#pragma unroll
    for (int b = 0; b < 2; ++b) {
        dstA0[b] = lds_off32(&lA[b][rowA * 64 + (((segA * 2 + 0) ^ swzA) * 16)]);
        dstA1[b] = lds_off32(&lA[b][rowA * 64 + (((segA * 2 + 1) ^ swzA) * 16)]);
        dstB0[b] = lds_off32(&lB[b][rowA * 64 + (((segA * 2 + 0) ^ swzB) * 16)]);
        dstB1[b] = lds_off32(&lB[b][rowA * 64 + (((segA * 2 + 1) ^ swzB) * 16)]);
    }
    const uint64_t baseA = (uint64_t)(uintptr_t)xq;
    const uint64_t baseB = (uint64_t)(uintptr_t)wq;
    const uint32_t voffA = (uint32_t)(t0 + rowA) * (uint32_t)D + (uint32_t)segA * 32u;
    const uint32_t voffB = (uint32_t)(n0 + rowA) * (uint32_t)D + (uint32_t)segA * 32u;

    async_copy_b128(dstA0[0], baseA, voffA);
    async_copy_b128(dstA1[0], baseA, voffA + 16u);
    async_copy_b128(dstB0[0], baseB, voffB);
    async_copy_b128(dstB1[0], baseB, voffB + 16u);

    if (tid < 128) ldq[tid] = dq[t0 + tid];

    v8i acc[2][4] = {};
    const int lr = lane & 15;
    const int hi = lane >> 4;

    wait_async0();
    __syncthreads();

    const int steps = D >> 6;
    for (int s = 0; s < steps; ++s) {
        const int cur = s & 1;
        const int nxt = cur ^ 1;

        if (s + 1 < steps) {
            const uint32_t ka = voffA + (uint32_t)(s + 1) * 64u;
            const uint32_t kb = voffB + (uint32_t)(s + 1) * 64u;
            async_copy_b128(dstA0[nxt], baseA, ka);
            async_copy_b128(dstA1[nxt], baseA, ka + 16u);
            async_copy_b128(dstB0[nxt], baseB, kb);
            async_copy_b128(dstB1[nxt], baseB, kb + 16u);
        }

        v8i afrag[2];
#pragma unroll
        for (int i = 0; i < 2; ++i) {
            const int am = wm + i * 16 + lr;
            const signed char* ar = &lA[cur][am * 64];
            const int e2 = (am >> 2) & 3;
#pragma unroll
            for (int p = 0; p < 4; ++p) {
                uint2 d = *(const uint2*)(ar + (((((p >> 1) * 2) + hi) ^ e2) * 16) + (p & 1) * 8);
                afrag[i][2 * p] = (int)d.x;
                afrag[i][2 * p + 1] = (int)d.y;
            }
        }
#pragma unroll
        for (int j = 0; j < 4; ++j) {
            const int bn = wn + j * 16 + lr;
            const signed char* br = &lB[cur][bn * 64];
            const int bs = ((bn >> 2) & 1) << 1;
            v8i bfrag;
#pragma unroll
            for (int h = 0; h < 2; ++h) {
                int4 d = *(const int4*)(br + (((h * 2 + hi) ^ bs) * 16));
                bfrag[4 * h + 0] = d.x;
                bfrag[4 * h + 1] = d.y;
                bfrag[4 * h + 2] = d.z;
                bfrag[4 * h + 3] = d.w;
            }
#pragma unroll
            for (int i = 0; i < 2; ++i) {
                acc[i][j] = __builtin_amdgcn_wmma_i32_16x16x64_iu8(
                    true, afrag[i], true, bfrag, acc[i][j], false, false);
            }
        }

        if (s + 1 < steps) wait_async0();
        __syncthreads();
    }

#pragma unroll
    for (int i = 0; i < 2; ++i) {
#pragma unroll
        for (int j = 0; j < 4; ++j) {
            const int n = n0 + wn + j * 16 + lr;
#pragma unroll
            for (int r = 0; r < 8; ++r) {
                const int mloc = wm + i * 16 + r + hi * 8;
                out[(size_t)(t0 + mloc) * N + n] = (float)acc[i][j][r] * ldq[mloc];
            }
        }
    }
}

extern "C" void kernel_launch(void* const* d_in, const int* in_sizes, int n_in,
                              void* d_out, int out_size, void* d_ws, size_t ws_size,
                              hipStream_t stream) {
    (void)n_in; (void)out_size; (void)ws_size;
    const float* x = (const float*)d_in[0];
    const float* w = (const float*)d_in[1];
    const float* g = (const float*)d_in[2];
    float* out = (float*)d_out;

    const int D = in_sizes[2];
    const int T = in_sizes[0] / D;
    const int N = in_sizes[1] / D;

    char* ws = (char*)d_ws;
    float* partials = (float*)ws;
    float* scalars = (float*)(ws + 4096);
    signed char* xq = (signed char*)(ws + 8192);
    signed char* wqp = xq + (size_t)T * D;
    float* dq = (float*)(wqp + (size_t)N * D);

    const int wElems4 = (N / 4) * D;

    k_abs_partial<<<1024, 256, 0, stream>>>(w, partials, wElems4);
    k_wscale<<<1, 256, 0, stream>>>(partials, scalars, 1024, 1.0f / ((float)N * (float)D));
    k_quant_w<<<4096, 256, 0, stream>>>(w, scalars, wqp, wElems4);
    k_quant_x<<<T, 256, 0, stream>>>(x, g, scalars, xq, dq, D);

    dim3 grid(N / 128, T / 128);
    k_gemm<<<grid, 256, 0, stream>>>(xq, wqp, dq, out, D, N);
}
